// QLoRABigNet_34239479284374
// MI455X (gfx1250) — matrix operand order repaired and checked by reading, not compiled, not run on hardware
//
#include <hip/hip_runtime.h>
#include <hip/hip_bf16.h>

typedef _Float16 v16h __attribute__((ext_vector_type(16)));
typedef _Float16 v8h  __attribute__((ext_vector_type(8)));
typedef _Float16 v4h  __attribute__((ext_vector_type(4)));
typedef float    v8f  __attribute__((ext_vector_type(8)));
typedef int      v4i  __attribute__((ext_vector_type(4)));

typedef __attribute__((address_space(1))) v4i gv4i;   // global int4
typedef __attribute__((address_space(3))) v4i lv4i;   // LDS int4

#define DIM   1024
#define BATCH 8192
#define RANK  32
#define NLIN  15

// ---------------------------------------------------------------------------
// Dequant: w = (f16(q) * f16(2/15) - 1) * group_scale   (f16 math, GROUP=16)
// ---------------------------------------------------------------------------
__global__ __launch_bounds__(256) void dequant4(const int* __restrict__ q,
                                                const _Float16* __restrict__ s,
                                                _Float16* __restrict__ W) {
    size_t idx = ((size_t)blockIdx.x * 256 + threadIdx.x) * 4;
    int4 qv = *(const int4*)(q + idx);
    _Float16 sc = s[idx >> 4];
    const _Float16 k   = (_Float16)(2.0f / 15.0f);
    const _Float16 one = (_Float16)1.0f;
    v4h w;
    w[0] = ((_Float16)(float)qv.x * k - one) * sc;
    w[1] = ((_Float16)(float)qv.y * k - one) * sc;
    w[2] = ((_Float16)(float)qv.z * k - one) * sc;
    w[3] = ((_Float16)(float)qv.w * k - one) * sc;
    *(v4h*)(W + idx) = w;
}

// ---------------------------------------------------------------------------
// f32 -> f16 shadow convert (initial x only; later shadows come from epilogues)
// ---------------------------------------------------------------------------
__global__ __launch_bounds__(256) void cvt_h(const float* __restrict__ X,
                                             _Float16* __restrict__ Xh) {
    size_t idx = ((size_t)blockIdx.x * 256 + threadIdx.x) * 4;
    float4 v = *(const float4*)(X + idx);
    v4h h;
    h[0] = (_Float16)v.x; h[1] = (_Float16)v.y;
    h[2] = (_Float16)v.z; h[3] = (_Float16)v.w;
    *(v4h*)(Xh + idx) = h;
}

// ---------------------------------------------------------------------------
// LoRA stage 1: T[m, r] = sum_k X[m,k] * A[r,k]   (f32, rank 32)
// ---------------------------------------------------------------------------
__global__ __launch_bounds__(256) void lora_xa(const float* __restrict__ X,
                                               const float* __restrict__ A,
                                               float* __restrict__ T) {
    int warp = threadIdx.x >> 5;
    int lane = threadIdx.x & 31;
    int m = blockIdx.x * 8 + warp;
    const float4* xr = (const float4*)(X + (size_t)m * DIM);
    const float4* ar = (const float4*)(A + (size_t)lane * DIM);
    float acc = 0.0f;
#pragma unroll 4
    for (int k = 0; k < DIM / 4; ++k) {
        float4 a = ar[k];
        float4 x = xr[k];
        acc += x.x * a.x + x.y * a.y + x.z * a.z + x.w * a.w;
    }
    T[(size_t)m * RANK + lane] = acc;
}

// ---------------------------------------------------------------------------
// Fused QLoRA linear, WMMA + async global->LDS double-buffered pipeline.
// Block tile 128(M) x 128(N), 256 threads = 8 waves (4 x 2), wave tile 32x64:
// 2 A-frags x 4 B-frags = 8 WMMA per 32-deep K step.
//   mode 0: out = ReLU(.)        mode 1: out = . + res[m,n]
// Writes f32 Out and (optionally) an f16 shadow OutH for the next GEMM.
// ---------------------------------------------------------------------------
__global__ __launch_bounds__(256) void qgemm_fused(const _Float16* __restrict__ Xh,
                                                   const _Float16* __restrict__ Wl,
                                                   const _Float16* __restrict__ bias,
                                                   const float*    __restrict__ T,
                                                   const float*    __restrict__ Bl,
                                                   const float*    __restrict__ res,
                                                   float*          __restrict__ Out,
                                                   _Float16*       __restrict__ OutH,
                                                   int mode) {
    __shared__ _Float16 Xs[2][128][32] __attribute__((aligned(32)));  // 16 KB
    __shared__ _Float16 Ws[2][128][32] __attribute__((aligned(32)));  // 16 KB
    __shared__ float    Bls[128][32]   __attribute__((aligned(16)));  // 16 KB
    __shared__ float    Ts[128][32]    __attribute__((aligned(16)));  // 16 KB

    const int t     = threadIdx.x;
    const int wave  = t >> 5;
    const int lane  = t & 31;
    const int lg    = lane >> 4;          // lane group 0/1
    const int lr    = lane & 15;
    const int wm    = wave >> 1;          // 0..3  -> M rows wm*32..+31
    const int wn    = wave & 1;           // 0..1  -> N cols wn*64..+63
    const int m_blk = blockIdx.y * 128;
    const int n_blk = blockIdx.x * 128;

    // ---- prologue: async-stage epilogue operands (Bl tile, T tile) ----
#pragma unroll
    for (int i = 0; i < 4; ++i) {                 // Bls: 128x32 f32, 16B/lane
        int id = t + i * 256;
        int row = id >> 3, c = id & 7;
        __builtin_amdgcn_global_load_async_to_lds_b128(
            (gv4i*)(Bl + (size_t)(n_blk + row) * RANK + c * 4),
            (lv4i*)&Bls[row][c * 4], 0, 0);
    }
#pragma unroll
    for (int i = 0; i < 4; ++i) {                 // Ts: 128x32 f32
        int id = t + i * 256;
        int row = id >> 3, c = id & 7;
        __builtin_amdgcn_global_load_async_to_lds_b128(
            (gv4i*)(T + (size_t)(m_blk + row) * RANK + c * 4),
            (lv4i*)&Ts[row][c * 4], 0, 0);
    }

    // ---- async tile copy: X/W tiles are 128x32 f16 = 512 x 16B chunks ----
    auto issue_tiles = [&](int p, int kb) {
#pragma unroll
        for (int i = 0; i < 2; ++i) {
            int id = t + i * 256;
            int row = id >> 2, c = id & 3;
            __builtin_amdgcn_global_load_async_to_lds_b128(
                (gv4i*)(Xh + (size_t)(m_blk + row) * DIM + kb + c * 8),
                (lv4i*)&Xs[p][row][c * 8], 0, 0);
        }
#pragma unroll
        for (int i = 0; i < 2; ++i) {
            int id = t + i * 256;
            int row = id >> 2, c = id & 3;
            __builtin_amdgcn_global_load_async_to_lds_b128(
                (gv4i*)(Wl + (size_t)(n_blk + row) * DIM + kb + c * 8),
                (lv4i*)&Ws[p][row][c * 8], 0, 0);
        }
    };

    issue_tiles(0, 0);

    v8f acc[2][4] = {};

    for (int kt = 0; kt < DIM / 32; ++kt) {
        const int p = kt & 1;
        __builtin_amdgcn_s_wait_asynccnt(0);   // buffer p (and prologue) landed
        __syncthreads();                       // all waves done with buffer p^1
        if (kt + 1 < DIM / 32) issue_tiles(p ^ 1, (kt + 1) * 32);

        // A fragments: per lane K = {b..b+15}, b = lg*16 (contiguous, matches B)
        const int base = lg * 16;
        v16h a[2];
#pragma unroll
        for (int mf = 0; mf < 2; ++mf) {
            const _Float16* xr = &Xs[p][wm * 32 + mf * 16 + lr][0];
            v8h alo = *(const v8h*)(xr + base);
            v8h ahi = *(const v8h*)(xr + base + 8);
            a[mf] = __builtin_shufflevector(alo, ahi, 0, 1, 2, 3, 4, 5, 6, 7,
                                            8, 9, 10, 11, 12, 13, 14, 15);
        }
        // B fragments: lane -> N = lr, K = lg*16 + h (contiguous 32B per lane)
#pragma unroll
        for (int nt = 0; nt < 4; ++nt) {
            v16h b = *(const v16h*)&Ws[p][wn * 64 + nt * 16 + lr][lg * 16];
#pragma unroll
            for (int mf = 0; mf < 2; ++mf) {
                acc[mf][nt] = __builtin_amdgcn_wmma_f32_16x16x32_f16(
                    false, a[mf], false, b, (short)0, acc[mf][nt], false, false);
            }
        }
    }

    // ---- epilogue: bias + 0.125 * (T @ Bl^T), ReLU or residual, dual store ----
    float bias_f[4];
#pragma unroll
    for (int nt = 0; nt < 4; ++nt)
        bias_f[nt] = (float)bias[n_blk + wn * 64 + nt * 16 + lr];

#pragma unroll
    for (int mf = 0; mf < 2; ++mf) {
#pragma unroll
        for (int r = 0; r < 8; ++r) {
            int mloc = wm * 32 + mf * 16 + lg * 8 + r;  // D layout: M = r + 8*lg
            int m = m_blk + mloc;
            float l0 = 0.f, l1 = 0.f, l2 = 0.f, l3 = 0.f;
#pragma unroll 8
            for (int rr = 0; rr < RANK; ++rr) {
                float tv = Ts[mloc][rr];
                l0 += tv * Bls[wn * 64 + lr][rr];
                l1 += tv * Bls[wn * 64 + 16 + lr][rr];
                l2 += tv * Bls[wn * 64 + 32 + lr][rr];
                l3 += tv * Bls[wn * 64 + 48 + lr][rr];
            }
            float lora[4] = {l0, l1, l2, l3};
#pragma unroll
            for (int nt = 0; nt < 4; ++nt) {
                int n = n_blk + wn * 64 + nt * 16 + lr;
                float val = acc[mf][nt][r] + bias_f[nt] + 0.125f * lora[nt];
                if (mode == 0) {
                    val = fmaxf(val, 0.0f);
                } else {
                    val += res[(size_t)m * DIM + n];
                }
                Out[(size_t)m * DIM + n] = val;
                if (OutH) OutH[(size_t)m * DIM + n] = (_Float16)val;
            }
        }
    }
}

// ---------------------------------------------------------------------------
// Row LayerNorm (in place) + f16 shadow write.
// ---------------------------------------------------------------------------
__global__ __launch_bounds__(256) void ln_rows(float* __restrict__ h,
                                               const float* __restrict__ g,
                                               const float* __restrict__ b,
                                               _Float16* __restrict__ outh) {
    __shared__ float red[256];
    const int t = threadIdx.x;
    float* p = h + (size_t)blockIdx.x * DIM;
    float4 v = ((const float4*)p)[t];

    red[t] = v.x + v.y + v.z + v.w;
    __syncthreads();
#pragma unroll
    for (int off = 128; off > 0; off >>= 1) {
        if (t < off) red[t] += red[t + off];
        __syncthreads();
    }
    float mu = red[0] * (1.0f / DIM);
    __syncthreads();

    float dx = v.x - mu, dy = v.y - mu, dz = v.z - mu, dw = v.w - mu;
    red[t] = dx * dx + dy * dy + dz * dz + dw * dw;
    __syncthreads();
#pragma unroll
    for (int off = 128; off > 0; off >>= 1) {
        if (t < off) red[t] += red[t + off];
        __syncthreads();
    }
    float rs = rsqrtf(red[0] * (1.0f / DIM) + 1e-5f);

    float4 gv = ((const float4*)g)[t];
    float4 bv = ((const float4*)b)[t];
    float4 o;
    o.x = dx * rs * gv.x + bv.x;
    o.y = dy * rs * gv.y + bv.y;
    o.z = dz * rs * gv.z + bv.z;
    o.w = dw * rs * gv.w + bv.w;
    ((float4*)p)[t] = o;

    v4h oh;
    oh[0] = (_Float16)o.x; oh[1] = (_Float16)o.y;
    oh[2] = (_Float16)o.z; oh[3] = (_Float16)o.w;
    ((v4h*)(outh + (size_t)blockIdx.x * DIM))[t] = oh;
}

// ---------------------------------------------------------------------------
extern "C" void kernel_launch(void* const* d_in, const int* in_sizes, int n_in,
                              void* d_out, int out_size, void* d_ws, size_t ws_size,
                              hipStream_t stream) {
    const float*    x      = (const float*)d_in[0];
    const int*      q_w    = (const int*)d_in[1];
    const _Float16* scales = (const _Float16*)d_in[2];
    const _Float16* biases = (const _Float16*)d_in[3];
    const float*    lora_A = (const float*)d_in[4];
    const float*    lora_B = (const float*)d_in[5];
    const float*    ln_g   = (const float*)d_in[6];
    const float*    ln_b   = (const float*)d_in[7];

    char* ws = (char*)d_ws;
    _Float16* Wq = (_Float16*)ws;                                       // 30 MiB
    size_t off = (size_t)NLIN * DIM * DIM * sizeof(_Float16);
    float* T  = (float*)(ws + off);  off += (size_t)BATCH * RANK * 4;   // 1 MiB
    float* hb = (float*)(ws + off);  off += (size_t)BATCH * DIM * 4;    // f32 act
    float* r1 = (float*)(ws + off);  off += (size_t)BATCH * DIM * 4;
    float* r2 = (float*)(ws + off);  off += (size_t)BATCH * DIM * 4;
    _Float16* F[2];
    F[0] = (_Float16*)(ws + off);    off += (size_t)BATCH * DIM * 2;    // f16 ping
    F[1] = (_Float16*)(ws + off);                                       // f16 pong

    dequant4<<<(NLIN * DIM * DIM) / (256 * 4), 256, 0, stream>>>(q_w, scales, Wq);
    cvt_h<<<(BATCH * DIM) / (256 * 4), 256, 0, stream>>>(x, F[0]);

    const dim3 ggrid(DIM / 128, BATCH / 128);   // (8, 64)
    const dim3 gblk(256);

    int li = 0, cur = 0;
    for (int blk = 0; blk < 6; ++blk) {
        const float* in0 = (blk == 0) ? x : hb;

        // j = 0 : r1 = ReLU(linear(in0))
        lora_xa<<<BATCH / 8, 256, 0, stream>>>(in0, lora_A + (size_t)li * RANK * DIM, T);
        qgemm_fused<<<ggrid, gblk, 0, stream>>>(F[cur], Wq + (size_t)li * DIM * DIM,
                                                biases + (size_t)li * DIM, T,
                                                lora_B + (size_t)li * DIM * RANK,
                                                nullptr, r1, F[cur ^ 1], 0);
        cur ^= 1; ++li;

        // j = 1 : r2 = ReLU(linear(r1))
        lora_xa<<<BATCH / 8, 256, 0, stream>>>(r1, lora_A + (size_t)li * RANK * DIM, T);
        qgemm_fused<<<ggrid, gblk, 0, stream>>>(F[cur], Wq + (size_t)li * DIM * DIM,
                                                biases + (size_t)li * DIM, T,
                                                lora_B + (size_t)li * DIM * RANK,
                                                nullptr, r2, F[cur ^ 1], 0);
        cur ^= 1; ++li;

        // j = 2 : h = linear(r2) + residual(in0)
        float* outp = (blk == 5) ? (float*)d_out : hb;
        lora_xa<<<BATCH / 8, 256, 0, stream>>>(r2, lora_A + (size_t)li * RANK * DIM, T);
        qgemm_fused<<<ggrid, gblk, 0, stream>>>(F[cur], Wq + (size_t)li * DIM * DIM,
                                                biases + (size_t)li * DIM, T,
                                                lora_B + (size_t)li * DIM * RANK,
                                                in0, outp, nullptr, 1);
        ++li;

        if (blk < 5) {
            ln_rows<<<BATCH, 256, 0, stream>>>(hb, ln_g + (size_t)blk * DIM,
                                               ln_b + (size_t)blk * DIM, F[cur ^ 1]);
            cur ^= 1;
        }
    }
    (void)in_sizes; (void)n_in; (void)out_size; (void)ws_size;
}
